// Bi_LSTM_Model_89386859364652
// MI455X (gfx1250) — matrix operand order repaired and checked by reading, not compiled, not run on hardware
//
#include <hip/hip_runtime.h>
#include <hip/hip_bf16.h>

typedef __attribute__((ext_vector_type(16))) __bf16 v16bf;
typedef __attribute__((ext_vector_type(8)))  __bf16 v8bf;
typedef __attribute__((ext_vector_type(8)))  float  v8f;

#define B_ROWS 128
#define T_STEPS 128
#define UDIM 512
#define G4 2048          // 4 * U
#define EMBP 320         // embedding padded to multiple of 32
#define EMBR 300

// ---------------------------------------------------------------------------
// Embedding gather: x[t][b][0:300] = emb[ids[b][t]]; cols 300..319 zero. bf16.
// ---------------------------------------------------------------------------
__global__ void gather_embed(const int* __restrict__ ids,
                             const float* __restrict__ emb,
                             __bf16* __restrict__ xb) {
  int idx = blockIdx.x * 256 + threadIdx.x;
  if (idx >= T_STEPS * B_ROWS * EMBP) return;
  int col = idx % EMBP;
  int tb  = idx / EMBP;
  int t = tb / B_ROWS, b = tb % B_ROWS;
  float v = 0.f;
  if (col < EMBR) {
    int id = ids[b * T_STEPS + t];
    v = emb[(size_t)id * EMBR + col];
  }
  xb[idx] = (__bf16)v;
}

// ---------------------------------------------------------------------------
// Pack [Wpad ; U] (K x 2048 f32) into WMMA B-fragment-native bf16 layout:
// flat[(((ntile*Kc + kk)*32 + lane)*16) + i] = M[K = kk*32 + (lane&16?16:0)+i]
//                                               [col = ntile*16 + (lane&15)]
// so each lane's 16 fragment values are one contiguous 32B load.
// ---------------------------------------------------------------------------
__global__ void pack_weights(const float* __restrict__ W,
                             const float* __restrict__ Uw,
                             __bf16* __restrict__ out,
                             int Kx, int Kc, int rowsW) {
  int idx = blockIdx.x * 256 + threadIdx.x;
  int total = 128 * Kc * 32 * 16;
  if (idx >= total) return;
  int i     = idx & 15;
  int lane  = (idx >> 4) & 31;
  int t2    = idx >> 9;
  int kk    = t2 % Kc;
  int ntile = t2 / Kc;
  int K   = kk * 32 + ((lane & 16) ? 16 : 0) + i;
  int col = ntile * 16 + (lane & 15);
  float v;
  if (K < Kx) v = (K < rowsW) ? W[(size_t)K * G4 + col] : 0.f;
  else        v = Uw[(size_t)(K - Kx) * G4 + col];
  out[idx] = (__bf16)v;
}

// ---------------------------------------------------------------------------
// Zero h (both phases, f32 + bf16 mirror) and c.
// ---------------------------------------------------------------------------
__global__ void init_state(float* __restrict__ hf, __bf16* __restrict__ hbb,
                           float* __restrict__ cf) {
  int idx = blockIdx.x * 256 + threadIdx.x;
  if (idx < 2 * 6 * B_ROWS * UDIM) { hf[idx] = 0.f; hbb[idx] = (__bf16)0.f; }
  if (idx < 6 * B_ROWS * UDIM) cf[idx] = 0.f;
}

__device__ __forceinline__ float sigf(float x)  { return 1.f / (1.f + __expf(-x)); }
__device__ __forceinline__ float ssign(float x) { return x / (1.f + fabsf(x)); }

// 16-bit A 16x32 fragment: idx0..15 -> K = half*16+idx (matches packed B)
__device__ __forceinline__ v16bf loadA(const __bf16* p, int hl8) {
  union { v16bf v; v8bf h2[2]; } u;
  u.h2[0] = *(const v8bf*)(p + 2 * hl8);
  u.h2[1] = *(const v8bf*)(p + 2 * hl8 + 8);
  return u.v;
}

// ---------------------------------------------------------------------------
// One LSTM timestep for one layer, both directions.
// grid = 512 blocks (dir[2] x mt[8] x nu[32]), block = 128 threads = 4 waves.
// Wave w computes gate w's 16x16 z-tile:  z = [x_t | h_prev] @ [W;U] + b
// via v_wmma_f32_16x16x32_bf16 with compile-time K split (no runtime branch
// in the inner loop -> immediate-offset b128 loads), two accumulators to
// halve the WMMA dependency chain, bias folded into the acc init.
// ---------------------------------------------------------------------------
template<int KCX, int KCH, int XP>
__global__ void __launch_bounds__(128)
lstm_step(const __bf16* __restrict__ xsrc_f, const __bf16* __restrict__ xsrc_b,
          const __bf16* __restrict__ pw_f, const __bf16* __restrict__ pw_b,
          const float* __restrict__ bias_f, const float* __restrict__ bias_b,
          float* __restrict__ hf_base, __bf16* __restrict__ hb_base,
          float* __restrict__ cf_base,
          int layer, int t)
{
  int bid = blockIdx.x;
  int dir = bid >> 8;          // 0 = fwd, 1 = bwd
  int rem = bid & 255;
  int mt  = rem >> 5;          // batch-row tile 0..7
  int nu  = rem & 31;          // h-column tile 0..31
  int w    = threadIdx.x >> 5; // wave id == gate (i,f,g,o)
  int lane = threadIdx.x & 31;

  const __bf16* xsrc = dir ? xsrc_b : xsrc_f;
  const __bf16* pw   = dir ? pw_b   : pw_f;
  const float*  bias = dir ? bias_b : bias_f;

  int sidx = dir * 3 + layer;
  int pin  = t & 1;
  int pout = pin ^ 1;
  const __bf16* hsrc = hb_base + ((size_t)pin * 6 + sidx) * (B_ROWS * UDIM);

  int ntile = w * 32 + nu;               // z column tile in [0,128)
  int row   = mt * 16 + (lane & 15);     // A-matrix row for this lane
  int hl8   = (lane >> 4) * 8;           // lane-half K offset (16-bit A layout)

  const __bf16* xp  = xsrc + (size_t)row * XP;
  const __bf16* hp  = hsrc + (size_t)row * UDIM;
  const __bf16* pwt = pw + ((size_t)ntile * (KCX + KCH)) * 512 + lane * 16;

  // bias is per-column; same value for all 8 row-VGPRs of the C/D tile
  float bv = bias[ntile * 16 + (lane & 15)];
  v8f acc0, acc1 = {};
  #pragma unroll
  for (int r = 0; r < 8; ++r) acc0[r] = bv;

  #pragma unroll
  for (int j = 0; j < KCX; ++j) {
    v16bf a = loadA(xp + j * 32, hl8);
    v16bf b = *(const v16bf*)(pwt + j * 512);
    if (j & 1) acc1 = __builtin_amdgcn_wmma_f32_16x16x32_bf16(false, a, false, b,
                                                              (short)0, acc1, false, false);
    else       acc0 = __builtin_amdgcn_wmma_f32_16x16x32_bf16(false, a, false, b,
                                                              (short)0, acc0, false, false);
  }
  #pragma unroll
  for (int j = 0; j < KCH; ++j) {
    v16bf a = loadA(hp + j * 32, hl8);
    v16bf b = *(const v16bf*)(pwt + (KCX + j) * 512);
    if ((KCX + j) & 1) acc1 = __builtin_amdgcn_wmma_f32_16x16x32_bf16(false, a, false, b,
                                                                      (short)0, acc1, false, false);
    else               acc0 = __builtin_amdgcn_wmma_f32_16x16x32_bf16(false, a, false, b,
                                                                      (short)0, acc0, false, false);
  }
  #pragma unroll
  for (int r = 0; r < 8; ++r) acc0[r] += acc1[r];

  // Spill the 4 gate tiles to LDS (C/D layout: VGPR r holds M = r + half*8, N = lane&15)
  __shared__ float zt[4][16][16];
  {
    int m0 = (lane >> 4) * 8;
    int n  = lane & 15;
    #pragma unroll
    for (int r = 0; r < 8; ++r) zt[w][m0 + r][n] = acc0[r];
  }
  __syncthreads();

  // Fused LSTM cell update (256 elements, 128 threads); bias already in z.
  for (int e = threadIdx.x; e < 256; e += 128) {
    int m = e >> 4, n = e & 15;
    int r    = mt * 16 + m;
    int hcol = nu * 16 + n;
    float zi = zt[0][m][n];
    float zf = zt[1][m][n];
    float zg = zt[2][m][n];
    float zo = zt[3][m][n];
    size_t coff = (size_t)sidx * (B_ROWS * UDIM) + (size_t)r * UDIM + hcol;
    size_t hoff = ((size_t)pout * 6 + sidx) * (B_ROWS * UDIM) + (size_t)r * UDIM + hcol;
    float c  = cf_base[coff];
    float cn = sigf(zf) * c + sigf(zi) * ssign(zg);
    float hn = sigf(zo) * ssign(cn);
    cf_base[coff] = cn;
    hf_base[hoff] = hn;
    hb_base[hoff] = (__bf16)hn;
  }
}

// ---------------------------------------------------------------------------
// Head: add = 0.5*(h_f + h_b); dense 512->256 + BN + PReLU; dense 256->7; softmax.
// One block per batch row, 256 threads (one per hidden col).
// ---------------------------------------------------------------------------
__global__ void __launch_bounds__(256)
head_kernel(const float* __restrict__ hff, const float* __restrict__ hfb,
            const float* __restrict__ d0W, const float* __restrict__ d0b,
            const float* __restrict__ gamma, const float* __restrict__ beta,
            const float* __restrict__ mean,  const float* __restrict__ var,
            const float* __restrict__ alpha,
            const float* __restrict__ d1W, const float* __restrict__ d1b,
            float* __restrict__ out)
{
  int row = blockIdx.x;
  int tid = threadIdx.x;
  __shared__ float hsh[256];
  __shared__ float sl[8];
  float s = 0.f;
  for (int k = 0; k < UDIM; ++k) {
    float a = 0.5f * (hff[(size_t)row * UDIM + k] + hfb[(size_t)row * UDIM + k]);
    s += a * d0W[(size_t)k * 256 + tid];
  }
  s += d0b[tid];
  s = (s - mean[tid]) * rsqrtf(var[tid] + 1e-3f) * gamma[tid] + beta[tid];
  s = s > 0.f ? s : alpha[tid] * s;
  hsh[tid] = s;
  __syncthreads();
  if (tid < 7) {
    float l = d1b[tid];
    for (int k = 0; k < 256; ++k) l += hsh[k] * d1W[k * 7 + tid];
    sl[tid] = l;
  }
  __syncthreads();
  if (tid == 0) {
    float mx = sl[0];
    for (int j = 1; j < 7; ++j) mx = fmaxf(mx, sl[j]);
    float ex[7]; float sum = 0.f;
    for (int j = 0; j < 7; ++j) { ex[j] = __expf(sl[j] - mx); sum += ex[j]; }
    for (int j = 0; j < 7; ++j) out[row * 7 + j] = ex[j] / sum;
  }
}

// ---------------------------------------------------------------------------
extern "C" void kernel_launch(void* const* d_in, const int* in_sizes, int n_in,
                              void* d_out, int out_size, void* d_ws, size_t ws_size,
                              hipStream_t stream) {
  (void)in_sizes; (void)n_in; (void)out_size; (void)ws_size;
  const int*   ids = (const int*)d_in[0];
  const float* emb = (const float*)d_in[1];
  const float *W[2][3], *Uw[2][3], *bias[2][3];
  for (int d = 0; d < 2; ++d)
    for (int l = 0; l < 3; ++l) {
      int base = 2 + d * 9 + l * 3;
      W[d][l]    = (const float*)d_in[base];
      Uw[d][l]   = (const float*)d_in[base + 1];
      bias[d][l] = (const float*)d_in[base + 2];
    }
  const float* d0W   = (const float*)d_in[20];
  const float* d0b   = (const float*)d_in[21];
  const float* gamma = (const float*)d_in[22];
  const float* beta  = (const float*)d_in[23];
  const float* mean  = (const float*)d_in[24];
  const float* var   = (const float*)d_in[25];
  const float* alpha = (const float*)d_in[26];
  const float* d1W   = (const float*)d_in[27];
  const float* d1b   = (const float*)d_in[28];

  static const int Kc_l[3]    = {26, 32, 32};
  static const int Kx_l[3]    = {320, 512, 512};
  static const int rowsW_l[3] = {300, 512, 512};

  char* ws = (char*)d_ws;
  size_t off = 0;
  __bf16* xb = (__bf16*)(ws + off);
  off += (size_t)T_STEPS * B_ROWS * EMBP * 2;                // 10.5 MB
  __bf16* pw[2][3];
  for (int d = 0; d < 2; ++d)
    for (int l = 0; l < 3; ++l) {
      pw[d][l] = (__bf16*)(ws + off);
      off += (size_t)128 * Kc_l[l] * 512 * 2;                // packed weights, ~23.6 MB total
    }
  float*  hf  = (float*)(ws + off);  off += (size_t)2 * 6 * B_ROWS * UDIM * 4;
  __bf16* hbb = (__bf16*)(ws + off); off += (size_t)2 * 6 * B_ROWS * UDIM * 2;
  float*  cf  = (float*)(ws + off);  off += (size_t)6 * B_ROWS * UDIM * 4;

  int nx = T_STEPS * B_ROWS * EMBP;
  gather_embed<<<(nx + 255) / 256, 256, 0, stream>>>(ids, emb, xb);
  for (int d = 0; d < 2; ++d)
    for (int l = 0; l < 3; ++l) {
      int tot = 128 * Kc_l[l] * 512;
      pack_weights<<<(tot + 255) / 256, 256, 0, stream>>>(
          W[d][l], Uw[d][l], pw[d][l], Kx_l[l], Kc_l[l], rowsW_l[l]);
    }
  init_state<<<(2 * 6 * B_ROWS * UDIM + 255) / 256, 256, 0, stream>>>(hf, hbb, cf);

  for (int t = 0; t < T_STEPS; ++t) {
    int pout = (t & 1) ^ 1;
    for (int l = 0; l < 3; ++l) {
      if (l == 0) {
        const __bf16* xsf = xb + (size_t)t * B_ROWS * EMBP;
        const __bf16* xsb = xb + (size_t)(T_STEPS - 1 - t) * B_ROWS * EMBP;
        lstm_step<10, 16, EMBP><<<512, 128, 0, stream>>>(
            xsf, xsb, pw[0][l], pw[1][l], bias[0][l], bias[1][l],
            hf, hbb, cf, l, t);
      } else {
        const __bf16* xsf = hbb + ((size_t)pout * 6 + 0 * 3 + (l - 1)) * (B_ROWS * UDIM);
        const __bf16* xsb = hbb + ((size_t)pout * 6 + 1 * 3 + (l - 1)) * (B_ROWS * UDIM);
        lstm_step<16, 16, UDIM><<<512, 128, 0, stream>>>(
            xsf, xsb, pw[0][l], pw[1][l], bias[0][l], bias[1][l],
            hf, hbb, cf, l, t);
      }
    }
  }
  // T=128 is even -> final h written to phase 0
  const float* hff = hf + ((size_t)0 * 6 + 2) * (B_ROWS * UDIM);
  const float* hfb = hf + ((size_t)0 * 6 + 5) * (B_ROWS * UDIM);
  head_kernel<<<128, 256, 0, stream>>>(hff, hfb, d0W, d0b, gamma, beta, mean, var,
                                       alpha, d1W, d1b, (float*)d_out);
}
